// BiologicalMultiHeadAttention_75187697484093
// MI455X (gfx1250) — matrix operand order repaired and checked by reading, not compiled, not run on hardware
//
#include <hip/hip_runtime.h>

// ---------------- problem constants ----------------
#define B_ 4
#define S_ 2048
#define E_ 1024
#define H_ 16
#define D_ 64
#define MROWS (B_*S_)   // 8192
#define E4_ (E_/4)      // 256

typedef __attribute__((ext_vector_type(16))) __bf16 v16bf;
typedef __attribute__((ext_vector_type(8)))  __bf16 v8bf;
typedef __attribute__((ext_vector_type(8)))  float  v8f;
typedef __attribute__((ext_vector_type(4)))  unsigned int u32x4;
typedef __attribute__((ext_vector_type(8)))  int i32x8;
typedef __attribute__((ext_vector_type(4)))  int i32x4;

#define AS1 __attribute__((address_space(1)))
#define AS3 __attribute__((address_space(3)))

// ---------------- CDNA5 feature detection ----------------
#if __has_builtin(__builtin_amdgcn_global_load_async_to_lds_b128)
#define HAVE_ASYNC 1
#else
#define HAVE_ASYNC 0
#endif

#if __has_builtin(__builtin_amdgcn_tensor_load_to_lds)
#define HAVE_TDM 1
#if __has_include(<hip/amd_detail/amd_gfx1250_TDM.h>)
// amdgpu-toolchain (therock): 6-arg form
#define TDM_LOAD(g0, g1) do { i32x4 z4_ = {}; i32x8 z8_ = {}; \
    __builtin_amdgcn_tensor_load_to_lds((g0), (g1), z4_, z4_, z8_, 0); } while (0)
#else
// ROCm 7.2: 5-arg form
#define TDM_LOAD(g0, g1) do { i32x4 z4_ = {}; \
    __builtin_amdgcn_tensor_load_to_lds((g0), (g1), z4_, z4_, 0); } while (0)
#endif
#else
#define HAVE_TDM 0
#endif

#if __has_builtin(__builtin_amdgcn_s_wait_asynccnt)
#define WAIT_ASYNC(n) __builtin_amdgcn_s_wait_asynccnt(n)
#else
#define WAIT_ASYNC(n) asm volatile("s_wait_asynccnt %0" :: "n"(n) : "memory")
#endif

#if __has_builtin(__builtin_amdgcn_s_wait_tensorcnt)
#define WAIT_TENSOR(n) __builtin_amdgcn_s_wait_tensorcnt(n)
#else
#define WAIT_TENSOR(n) asm volatile("s_wait_tensorcnt %0" :: "n"(n) : "memory")
#endif

__device__ __forceinline__ v16bf cat8(v8bf lo, v8bf hi) {
  return __builtin_shufflevector(lo, hi, 0,1,2,3,4,5,6,7,8,9,10,11,12,13,14,15);
}

__device__ __forceinline__ v8f wmma_bf16(v16bf a, v16bf b, v8f c) {
  return __builtin_amdgcn_wmma_f32_16x16x32_bf16(false, a, false, b, (short)0, c, false, false);
}

__device__ __forceinline__ unsigned lds_off(const void* p) {
  return (unsigned)(size_t)(AS3 const void*)p;
}

// ---- Tensor DMA descriptor (D#) builders, ISA cdna5 ch.8 ----
// group0: count=1 (user), lds_addr, global_addr, type=2
__device__ __forceinline__ u32x4 tdm_g0(unsigned lds_addr, const void* gaddr) {
  unsigned long long ga = (unsigned long long)(size_t)gaddr;
  u32x4 g;
  g[0] = 1u;                                   // count=1, is_restore=0, gather off
  g[1] = lds_addr;                             // LDS byte address
  g[2] = (unsigned)ga;                         // global_addr[31:0]
  g[3] = ((unsigned)(ga >> 32) & 0x01FFFFFFu)  // global_addr[56:32]
       | (2u << 30);                           // type=2 ("image")
  return g;
}
// group1: data_size=2B, 2D tile: tile_dim0 x tile_dim1, row stride = stride0 elems
__device__ __forceinline__ i32x8 tdm_g1(unsigned tdim0, unsigned tdim1,
                                        unsigned tile0, unsigned tile1,
                                        unsigned long long stride0) {
  unsigned long long q0 = (1ull << 16)                               // data_size: 1 -> 2 bytes
                        | ((unsigned long long)(tdim0 & 0xFFFFu) << 48);
  unsigned long long q1 = (unsigned long long)(tdim0 >> 16)
                        | ((unsigned long long)tdim1 << 16)
                        | ((unsigned long long)(tile0 & 0xFFFFu) << 48);
  unsigned long long q2 = (unsigned long long)(tile1 & 0xFFFFu)      // tile_dim1
                        | ((stride0 & 0xFFFFFFFFull) << 32);         // dim0 stride lo
  unsigned long long q3 = (stride0 >> 32) & 0xFFFFull;               // dim0 stride hi
  union { unsigned long long q[4]; i32x8 v; } u;
  u.q[0] = q0; u.q[1] = q1; u.q[2] = q2; u.q[3] = q3;
  return u.v;
}

// ---------------- fp32 -> bf16 cast ----------------
__global__ void cast_f32_bf16(const float* __restrict__ in, __bf16* __restrict__ out, int n) {
  int i = blockIdx.x * blockDim.x + threadIdx.x;
  int stride = gridDim.x * blockDim.x;
  for (; i < n; i += stride) out[i] = (__bf16)in[i];
}

// ---------------- bf16 WMMA GEMM, W tile staged in LDS (async, double-buffered) ----------------
// C[M,N] = A[M,K] @ W[N,K]^T + bias[N]; flags: bit0 = store bf16, bit1 = relu
// 256 thr = 8 waves; block tile 128(M) x 128(N); wave tile 32x64 (2x4 accs).
__global__ void __launch_bounds__(256)
gemm_bf16_wmma(const __bf16* __restrict__ A, const __bf16* __restrict__ W,
               const float* __restrict__ bias, void* __restrict__ out,
               int M, int N, int K, int flags)
{
  __shared__ __align__(16) __bf16 Wl[2][128][40];  // 32 k + 8 pad per row, 2 buffers (20 KB)

  const int tid  = threadIdx.x;
  const int lane = tid & 31;
  const int wave = tid >> 5;
  const int ln   = lane & 15;
  const int hf   = lane >> 4;
  const int wm   = wave & 3;                 // 4 waves along M
  const int wn   = wave >> 2;                // 2 waves along N
  const int m0   = blockIdx.y * 128 + wm * 32;
  const int nblk = blockIdx.x * 128;
  const int n0   = nblk + wn * 64;

  const __bf16* Ar0 = A + (size_t)(m0 + ln) * K;
  const __bf16* Ar1 = A + (size_t)(m0 + 16 + ln) * K;
  const int aoff = hf * 8;

  // stage one 128x32 W tile into LDS buffer (512 16-byte chunks, 2 per thread)
  auto stage = [&](int buf, int k) {
    #pragma unroll
    for (int i = 0; i < 2; ++i) {
      const int q   = tid + i * 256;
      const int row = q >> 2;
      const int ch  = q & 3;
      const __bf16* g = W + (size_t)(nblk + row) * K + k + ch * 8;
      __bf16* l = &Wl[buf][row][ch * 8];
#if HAVE_ASYNC
      __builtin_amdgcn_global_load_async_to_lds_b128((AS1 i32x4*)g, (AS3 i32x4*)l, 0, 0);
#else
      *(v8bf*)l = *(const v8bf*)g;
#endif
    }
  };

  v8f acc[2][4] = {};
  stage(0, 0);
  for (int k = 0; k < K; k += 32) {
    const int buf = (k >> 5) & 1;
    const bool next = (k + 32) < K;
    if (next) {
      stage(buf ^ 1, k + 32);
      __builtin_prefetch(Ar0 + k + 64, 0, 1);
    }
#if HAVE_ASYNC
    if (next) WAIT_ASYNC(2); else WAIT_ASYNC(0);  // retire current buffer, keep next in flight
#endif
    __syncthreads();

    v16bf a0 = cat8(*(const v8bf*)(Ar0 + k + aoff), *(const v8bf*)(Ar0 + k + aoff + 16));
    v16bf a1 = cat8(*(const v8bf*)(Ar1 + k + aoff), *(const v8bf*)(Ar1 + k + aoff + 16));
    #pragma unroll
    for (int jj = 0; jj < 4; ++jj) {
      const __bf16* bl = &Wl[buf][wn * 64 + jj * 16 + ln][hf * 8];
      v16bf b = cat8(*(const v8bf*)bl, *(const v8bf*)(bl + 16));
      acc[0][jj] = wmma_bf16(a0, b, acc[0][jj]);
      acc[1][jj] = wmma_bf16(a1, b, acc[1][jj]);
    }
    __syncthreads();
  }

  // C layout: VGPR r, lanes 0-15 -> row r, lanes 16-31 -> row r+8; col = lane&15
  #pragma unroll
  for (int i = 0; i < 2; ++i)
    #pragma unroll
    for (int jj = 0; jj < 4; ++jj) {
      const int col = n0 + jj * 16 + ln;
      const float bcol = bias[col];
      #pragma unroll
      for (int r = 0; r < 8; ++r) {
        const int row = m0 + i * 16 + r + hf * 8;
        float v = acc[i][jj][r] + bcol;
        if (flags & 2) v = fmaxf(v, 0.0f);
        if (flags & 1) ((__bf16*)out)[(size_t)row * N + col] = (__bf16)v;
        else           ((float*)out)[(size_t)row * N + col] = v;
      }
    }
}

// ---------------- pack V -> V^T [b,h,d,s] (bf16 gather) ----------------
__global__ void pack_vt(const __bf16* __restrict__ Vf, __bf16* __restrict__ Vt, int n) {
  int i = blockIdx.x * blockDim.x + threadIdx.x;
  int stride = gridDim.x * blockDim.x;
  for (; i < n; i += stride) {
    int s = i % S_;       int t = i / S_;
    int d = t % D_;       t /= D_;
    int h = t % H_;       int b = t / H_;
    Vt[i] = Vf[((size_t)(b * S_ + s)) * E_ + h * D_ + d];
  }
}

// ---------------- flash attention: TDM-staged K/V tiles + WMMA ----------------
// Qf,Kf: bf16 [b,s,(h,d)] (row stride E_); Vt: bf16 [b,h,d,s]; Oo: f32 [b,h,s,d]
__global__ void __launch_bounds__(128)
flash_attn_wmma(const __bf16* __restrict__ Qf, const __bf16* __restrict__ Kf,
                const __bf16* __restrict__ Vt, float* __restrict__ Oo)
{
  __shared__ __align__(16) __bf16 Kl[2][32][64];   // 8 KB, double-buffered K tile
  __shared__ __align__(16) __bf16 Vl[2][64][32];   // 8 KB, double-buffered V^T tile
  __shared__ __align__(16) __bf16 Pl[4][16][32];   // per-wave P relay (C-layout -> A-layout)

  const int tid  = threadIdx.x;
  const int lane = tid & 31;
  const int wave = tid >> 5;
  const int ln   = lane & 15;
  const int hf   = lane >> 4;

  const int nqt = S_ / 64;
  const int qt  = blockIdx.x % nqt;
  const int bh  = blockIdx.x / nqt;
  const int h   = bh % H_;
  const int b   = bh / H_;
  const int q0  = qt * 64 + wave * 16;

  // issue K (32x64, row stride E_) and V^T (64x32, row stride S_) tile loads
  auto stage = [&](int buf, int j) {
#if HAVE_TDM
    if (wave == 0) {
      const __bf16* kg = Kf + ((size_t)(b * S_ + j)) * E_ + h * D_;
      TDM_LOAD(tdm_g0(lds_off(&Kl[buf][0][0]), kg),
               tdm_g1(/*tdim0*/D_, /*tdim1*/S_, /*tile0*/D_, /*tile1*/32, /*stride0*/E_));
      const __bf16* vg = Vt + ((size_t)(b * H_ + h) * D_) * S_ + j;
      TDM_LOAD(tdm_g0(lds_off(&Vl[buf][0][0]), vg),
               tdm_g1(/*tdim0*/S_, /*tdim1*/D_, /*tile0*/32, /*tile1*/D_, /*stride0*/S_));
    }
#else
    for (int q = tid; q < 256; q += 128) {      // K: 32 rows x 8 chunks
      const int row = q >> 3, ch = q & 7;
      *(v8bf*)&Kl[buf][row][ch * 8] =
          *(const v8bf*)(Kf + ((size_t)(b * S_ + j + row)) * E_ + h * D_ + ch * 8);
    }
    for (int q = tid; q < 256; q += 128) {      // V: 64 rows x 4 chunks
      const int row = q >> 2, ch = q & 3;
      *(v8bf*)&Vl[buf][row][ch * 8] =
          *(const v8bf*)(Vt + ((size_t)((b * H_ + h) * D_ + row)) * S_ + j + ch * 8);
    }
#endif
  };

  // Q A-frags for d in [0,32) and [32,64) held in registers for the whole loop
  const __bf16* Qrow = Qf + ((size_t)(b * S_ + q0 + ln)) * E_ + h * D_;
  v16bf qa[2];
  #pragma unroll
  for (int db = 0; db < 2; ++db) {
    const int d = db * 32 + hf * 8;
    qa[db] = cat8(*(const v8bf*)(Qrow + d), *(const v8bf*)(Qrow + d + 16));
  }

  float mrow[8], lrow[8];
  #pragma unroll
  for (int r = 0; r < 8; ++r) { mrow[r] = -1e30f; lrow[r] = 0.0f; }
  v8f Oacc[4] = {};
  const float sm_scale = 0.125f;  // D^-0.5

  stage(0, 0);
  for (int j = 0; j < S_; j += 32) {
    const int buf = (j >> 5) & 1;
    const bool next = (j + 32) < S_;
    if (next) stage(buf ^ 1, j + 32);
#if HAVE_TDM
    if (wave == 0) { if (next) WAIT_TENSOR(2); else WAIT_TENSOR(0); }
#endif
    __syncthreads();

    // ---- scores: 16 q-rows x 32 keys from LDS ----
    v8f s[2] = {};
    #pragma unroll
    for (int t = 0; t < 2; ++t)
      #pragma unroll
      for (int db = 0; db < 2; ++db) {
        const __bf16* kp = &Kl[buf][t * 16 + ln][db * 32 + hf * 8];
        v16bf kb = cat8(*(const v8bf*)kp, *(const v8bf*)(kp + 16));
        s[t] = wmma_bf16(qa[db], kb, s[t]);
      }

    // ---- online softmax (row reductions across 16-lane N groups) ----
    float alpha[8];
    #pragma unroll
    for (int r = 0; r < 8; ++r) {
      float s0 = s[0][r] * sm_scale;
      float s1 = s[1][r] * sm_scale;
      float vm = fmaxf(s0, s1);
      for (int msk = 1; msk < 16; msk <<= 1) vm = fmaxf(vm, __shfl_xor(vm, msk, 32));
      const float nm = fmaxf(mrow[r], vm);
      alpha[r] = __expf(mrow[r] - nm);
      const float p0 = __expf(s0 - nm);
      const float p1 = __expf(s1 - nm);
      float rs = p0 + p1;
      for (int msk = 1; msk < 16; msk <<= 1) rs += __shfl_xor(rs, msk, 32);
      lrow[r] = lrow[r] * alpha[r] + rs;
      mrow[r] = nm;
      const int row = r + hf * 8;
      Pl[wave][row][ln]      = (__bf16)p0;
      Pl[wave][row][16 + ln] = (__bf16)p1;
    }
    #pragma unroll
    for (int dt = 0; dt < 4; ++dt)
      #pragma unroll
      for (int r = 0; r < 8; ++r) Oacc[dt][r] *= alpha[r];

    // ---- P (A-layout) from LDS, then PV from V^T tile ----
    v16bf pa = cat8(*(const v8bf*)&Pl[wave][ln][hf * 8],
                    *(const v8bf*)&Pl[wave][ln][16 + hf * 8]);
    #pragma unroll
    for (int dt = 0; dt < 4; ++dt) {
      const __bf16* vp = &Vl[buf][dt * 16 + ln][hf * 8];
      v16bf vb = cat8(*(const v8bf*)vp, *(const v8bf*)(vp + 16));
      Oacc[dt] = wmma_bf16(pa, vb, Oacc[dt]);
    }
    __syncthreads();
  }

  // ---- normalize + store f32 [b,h,s,d] ----
  #pragma unroll
  for (int dt = 0; dt < 4; ++dt)
    #pragma unroll
    for (int r = 0; r < 8; ++r) {
      const int srow = q0 + r + hf * 8;
      const int d = dt * 16 + ln;
      Oo[((size_t)(b * H_ + h) * S_ + srow) * D_ + d] = Oacc[dt][r] * (1.0f / lrow[r]);
    }
}

// ---------------- neuromod fuse ----------------
__global__ void fuse_mod(const float* __restrict__ Oo, const float* __restrict__ Mf,
                         const float* __restrict__ dop, const float* __restrict__ ser,
                         const float* __restrict__ nor, const float* __restrict__ ace,
                         const float* __restrict__ asc, const float* __restrict__ abi,
                         __bf16* __restrict__ Y, int n)
{
  const float gain = (dop[0] + ser[0] + nor[0] + ace[0]) * 0.25f;
  const float sc = asc[0], bi = abi[0];
  int i = blockIdx.x * blockDim.x + threadIdx.x;
  int stride = gridDim.x * blockDim.x;
  for (; i < n; i += stride) {
    const int e = i % E_;     int t = i / E_;
    const int s = t % S_;     const int b = t / S_;
    const int h = e / D_;     const int d = e % D_;
    const float attn = Oo[((size_t)(b * H_ + h) * S_ + s) * D_ + d];
    const float v = (attn * sc + bi) * (1.0f + Mf[i] * gain);
    Y[i] = (__bf16)v;
  }
}

// ---------------- host driver ----------------
extern "C" void kernel_launch(void* const* d_in, const int* in_sizes, int n_in,
                              void* d_out, int out_size, void* d_ws, size_t ws_size,
                              hipStream_t stream) {
  const float* query = (const float*)d_in[0];
  const float* Wq  = (const float*)d_in[1];
  const float* bq  = (const float*)d_in[2];
  const float* Wk  = (const float*)d_in[3];
  const float* bk  = (const float*)d_in[4];
  const float* Wv  = (const float*)d_in[5];
  const float* bv  = (const float*)d_in[6];
  const float* Wo  = (const float*)d_in[7];
  const float* bo  = (const float*)d_in[8];
  const float* Wm1 = (const float*)d_in[9];
  const float* bm1 = (const float*)d_in[10];
  const float* Wm2 = (const float*)d_in[11];
  const float* bm2 = (const float*)d_in[12];
  const float* dop = (const float*)d_in[13];
  const float* ser = (const float*)d_in[14];
  const float* nor = (const float*)d_in[15];
  const float* ace = (const float*)d_in[16];
  const float* asc = (const float*)d_in[17];
  const float* abi = (const float*)d_in[18];

  char* ws = (char*)d_ws;
  const size_t MB = (size_t)1 << 20;
  __bf16* Xbf  = (__bf16*)(ws + 0);        // 16 MB  [8192,1024]
  __bf16* Wqb  = (__bf16*)(ws + 16 * MB);
  __bf16* Wkb  = (__bf16*)(ws + 18 * MB);
  __bf16* Wvb  = (__bf16*)(ws + 20 * MB);
  __bf16* Wob  = (__bf16*)(ws + 22 * MB);
  __bf16* Wm1b = (__bf16*)(ws + 24 * MB);
  __bf16* Wm2b = (__bf16*)(ws + 25 * MB);
  __bf16* Qf   = (__bf16*)(ws + 32 * MB);
  __bf16* Kf   = (__bf16*)(ws + 48 * MB);
  __bf16* Vf   = (__bf16*)(ws + 64 * MB);
  __bf16* Vt   = (__bf16*)(ws + 80 * MB);
  float*  Oo   = (float*) (ws + 96 * MB);  // 32 MB
  __bf16* H1   = (__bf16*)(ws + 128 * MB); //  4 MB  [8192,256]
  float*  Mf   = (float*) (ws + 132 * MB); // 32 MB
  __bf16* Yb   = (__bf16*)(ws + 164 * MB); // 16 MB

  cast_f32_bf16<<<1024, 256, 0, stream>>>(query, Xbf, MROWS * E_);
  cast_f32_bf16<<<512, 256, 0, stream>>>(Wq,  Wqb,  E_ * E_);
  cast_f32_bf16<<<512, 256, 0, stream>>>(Wk,  Wkb,  E_ * E_);
  cast_f32_bf16<<<512, 256, 0, stream>>>(Wv,  Wvb,  E_ * E_);
  cast_f32_bf16<<<512, 256, 0, stream>>>(Wo,  Wob,  E_ * E_);
  cast_f32_bf16<<<256, 256, 0, stream>>>(Wm1, Wm1b, E4_ * E_);
  cast_f32_bf16<<<256, 256, 0, stream>>>(Wm2, Wm2b, E_ * E4_);

  const dim3 blk(256);
  const dim3 gEE(E_ / 128,  MROWS / 128);   // (8, 64)
  const dim3 gE4(E4_ / 128, MROWS / 128);   // (2, 64)

  gemm_bf16_wmma<<<gEE, blk, 0, stream>>>(Xbf, Wqb, bq, Qf, MROWS, E_, E_, 1);
  gemm_bf16_wmma<<<gEE, blk, 0, stream>>>(Xbf, Wkb, bk, Kf, MROWS, E_, E_, 1);
  gemm_bf16_wmma<<<gEE, blk, 0, stream>>>(Xbf, Wvb, bv, Vf, MROWS, E_, E_, 1);
  pack_vt<<<2048, 256, 0, stream>>>(Vf, Vt, B_ * H_ * D_ * S_);

  gemm_bf16_wmma<<<gE4, blk, 0, stream>>>(Xbf, Wm1b, bm1, H1, MROWS, E4_, E_, 1 | 2);
  gemm_bf16_wmma<<<gEE, blk, 0, stream>>>(H1, Wm2b, bm2, Mf, MROWS, E_, E4_, 0);

  flash_attn_wmma<<<B_ * H_ * (S_ / 64), 128, 0, stream>>>(Qf, Kf, Vt, Oo);

  fuse_mod<<<2048, 256, 0, stream>>>(Oo, Mf, dop, ser, nor, ace, asc, abi, Yb, B_ * S_ * E_);
  gemm_bf16_wmma<<<gEE, blk, 0, stream>>>(Yb, Wob, bo, (float*)d_out, MROWS, E_, E_, 0);
}
